// DeformableCrossAttention_37847251812561
// MI455X (gfx1250) — matrix operand order repaired and checked by reading, not compiled, not run on hardware
//
#include <hip/hip_runtime.h>
#include <hip/hip_bf16.h>

typedef __attribute__((ext_vector_type(16))) _Float16 v16h;
typedef __attribute__((ext_vector_type(8)))  _Float16 v8h;
typedef __attribute__((ext_vector_type(8)))  float    v8f;

#define NHEAD 8
#define NPTS  4
#define DDIM  256
#define KSTEPS 8   // 256 / 32

// ---------------------------------------------------------------------------
// Tensor Data Mover helpers (CDNA5 ISA-8)
// ---------------------------------------------------------------------------
#if __has_include(<hip/amd_detail/amd_gfx1250_TDM.h>)
#define TDM_6ARG 1
#else
#define TDM_6ARG 0
#endif

typedef __attribute__((ext_vector_type(4))) unsigned int tdm_u32x4;
typedef __attribute__((ext_vector_type(8))) int          tdm_i32x8;
typedef __attribute__((ext_vector_type(4))) int          tdm_i32x4;

// Contiguous 1-D f16 copy: nelem halves gsrc -> LDS (tile_dim0 = nelem <= 65535).
__device__ __forceinline__ void tdm_copy_f16_1d(const _Float16* gsrc, void* ldst,
                                                unsigned nelem, unsigned tensor_elems) {
  unsigned long long ga = (unsigned long long)gsrc;
  unsigned lds = (unsigned)(unsigned long long)ldst;   // flat LDS addr[31:0] = LDS offset
  tdm_u32x4 g0;
  g0.x = 1u;                                           // count=1 (valid), user mode
  g0.y = lds;                                          // lds_addr (bytes)
  g0.z = (unsigned)ga;                                 // global_addr[31:0]
  g0.w = ((unsigned)(ga >> 32) & 0x01FFFFFFu) | 0x80000000u;  // ga[56:32] | type=2
  tdm_i32x8 g1;
  g1[0] = 0x00010000;                                  // wg_mask=0, data_size=1 (2B)
  g1[1] = (int)((tensor_elems & 0xFFFFu) << 16);       // tensor_dim0[15:0]
  g1[2] = (int)((tensor_elems >> 16) & 0xFFFFu) | (1 << 16); // tensor_dim0[31:16] | tensor_dim1=1
  g1[3] = (int)((nelem & 0xFFFFu) << 16);              // tensor_dim1 hi=0 | tile_dim0
  g1[4] = 1;                                           // tile_dim1=1, tile_dim2=0
  g1[5] = (int)tensor_elems;                           // tensor_dim0_stride[31:0]
  g1[6] = 0;                                           // stride hi, dim1_stride lo
  g1[7] = 0;
  tdm_i32x4 z4 = {0, 0, 0, 0};
#if TDM_6ARG
  tdm_i32x8 z8 = {0, 0, 0, 0, 0, 0, 0, 0};
  __builtin_amdgcn_tensor_load_to_lds(g0, g1, z4, z4, z8, 0);
#else
  __builtin_amdgcn_tensor_load_to_lds(g0, g1, z4, z4, 0);
#endif
}

// ---------------------------------------------------------------------------
// WMMA helpers (CDNA5 wave32 layouts, CDNA5 ISA §7.12.2)
// ---------------------------------------------------------------------------
__device__ __forceinline__ v16h load_a_frag_lds(const _Float16* ldsA, int lane, int kstep) {
  const int r  = lane & 15;
  const int hi = lane >> 4;
  const _Float16* p0 = ldsA + r * DDIM + kstep * 32 + hi * 8;
  v8h lo  = *(const v8h*)(p0);
  v8h hi8 = *(const v8h*)(p0 + 16);
  return __builtin_shufflevector(lo, hi8, 0,1,2,3,4,5,6,7,8,9,10,11,12,13,14,15);
}

__device__ __forceinline__ v16h load_b_frag(const _Float16* __restrict__ packed,
                                            int ntiles, int kstep, int ntile, int lane) {
  const _Float16* p = packed + (((size_t)(kstep * ntiles + ntile)) << 9) + (lane << 4);
  return *(const v16h*)p;
}

__device__ __forceinline__ v8f wmma_f16(v16h a, v16h b, v8f c) {
  return __builtin_amdgcn_wmma_f32_16x16x32_f16(false, a, false, b, (short)0, c, false, false);
}

// ---------------------------------------------------------------------------
// Kernel 0a: fp32 -> f16 bulk convert (SADQ)
// ---------------------------------------------------------------------------
__global__ void f32_to_f16_kernel(const float* __restrict__ in, _Float16* __restrict__ out, int n4) {
  int i = blockIdx.x * blockDim.x + threadIdx.x;
  if (i >= n4) return;
  float4 v = ((const float4*)in)[i];
  _Float16* d = out + (size_t)i * 4;
  d[0] = (_Float16)v.x; d[1] = (_Float16)v.y;
  d[2] = (_Float16)v.z; d[3] = (_Float16)v.w;
}

// ---------------------------------------------------------------------------
// Kernel 0b: fp32 weight [256 x N] -> f16 B-fragment layout.
// ---------------------------------------------------------------------------
__global__ void pack_weights_kernel(const float* __restrict__ W, _Float16* __restrict__ out, int N) {
  int idx = blockIdx.x * blockDim.x + threadIdx.x;
  int total = DDIM * N;
  if (idx >= total) return;
  int j     = idx & 15;
  int lane  = (idx >> 4) & 31;
  int chunk = idx >> 9;
  int ntiles = N >> 4;
  int ntile  = chunk % ntiles;
  int kstep  = chunk / ntiles;
  int k   = kstep * 32 + (j >> 3) * 16 + (lane >> 4) * 8 + (j & 7);
  int col = ntile * 16 + (lane & 15);
  out[idx] = (_Float16)W[k * N + col];
}

// ---------------------------------------------------------------------------
// Kernel 1: projections + softmax + locs. One wave per 16-query tile.
// A-tile staged by the Tensor Data Mover.
// ---------------------------------------------------------------------------
__global__ void __launch_bounds__(32) proj_kernel(
    const _Float16* __restrict__ sadq_h, int totalQ,
    const _Float16* __restrict__ woP, const _Float16* __restrict__ waP,
    const float* __restrict__ bo, const float* __restrict__ ba,
    const float* __restrict__ Wr, const float* __restrict__ br,
    float* __restrict__ attn_out, float* __restrict__ locs_out) {
  __shared__ _Float16 ldsA[16 * DDIM];   // 8 KB f16 query tile
  __shared__ float    proj[16 * 96];     // cols 0..63 offsets, 64..95 attn logits
  __shared__ float    baseLoc[32];       // [16][2]

  const int lane = threadIdx.x;
  const int q0   = blockIdx.x * 16;

  // TDM: stage 16 rows x 256 halves (8 KB, contiguous) into LDS
  tdm_copy_f16_1d(sadq_h + (size_t)q0 * DDIM, ldsA, 16 * DDIM, (unsigned)totalQ * DDIM);
  __builtin_amdgcn_s_wait_tensorcnt(0);
  __syncthreads();

  // 6 WMMA output tiles: t=0..3 -> Wo (N=64), t=4..5 -> Wa (N=32)
  for (int t = 0; t < 6; ++t) {
    v8f acc = {};
    const _Float16* wptr = (t < 4) ? woP : waP;
    const int ntiles = (t < 4) ? 4 : 2;
    const int nt     = (t < 4) ? t : (t - 4);
    for (int ks = 0; ks < KSTEPS; ++ks) {
      v16h a = load_a_frag_lds(ldsA, lane, ks);
      v16h b = load_b_frag(wptr, ntiles, ks, nt, lane);
      acc = wmma_f16(a, b, acc);
    }
    const int col  = t * 16 + (lane & 15);
    const float bs = (t < 4) ? bo[col] : ba[col - 64];
    #pragma unroll
    for (int v = 0; v < 8; ++v) {
      int row = v + 8 * (lane >> 4);
      proj[row * 96 + col] = acc[v] + bs;
    }
  }

  // base reference (N=2): lane -> (row = lane&15, col = lane>>4)
  {
    const int r = lane & 15;
    const int c = lane >> 4;
    float dot = 0.f;
    for (int k = 0; k < DDIM; ++k)
      dot += (float)ldsA[r * DDIM + k] * Wr[k * 2 + c];
    baseLoc[r * 2 + c] = dot + br[c];
  }
  __syncthreads();

  // softmax over NPTS per (row, head) + loc = base + 0.1*offset
  for (int it = lane; it < 16 * NHEAD; it += 32) {
    const int r = it >> 3;
    const int h = it & 7;
    const float* lg = &proj[r * 96 + 64 + h * 4];
    float m  = fmaxf(fmaxf(lg[0], lg[1]), fmaxf(lg[2], lg[3]));
    float e0 = __expf(lg[0] - m), e1 = __expf(lg[1] - m);
    float e2 = __expf(lg[2] - m), e3 = __expf(lg[3] - m);
    float inv = 1.f / (e0 + e1 + e2 + e3);
    const int q = q0 + r;
    float* ao = attn_out + (size_t)q * 32 + h * 4;
    ao[0] = e0 * inv; ao[1] = e1 * inv; ao[2] = e2 * inv; ao[3] = e3 * inv;
    const float bx = baseLoc[r * 2 + 0];
    const float by = baseLoc[r * 2 + 1];
    float* lo = locs_out + (size_t)q * 64 + h * 8;
    #pragma unroll
    for (int p = 0; p < NPTS; ++p) {
      lo[p * 2 + 0] = bx + 0.1f * proj[r * 96 + h * 8 + p * 2 + 0];
      lo[p * 2 + 1] = by + 0.1f * proj[r * 96 + h * 8 + p * 2 + 1];
    }
  }
}

// ---------------------------------------------------------------------------
// Kernel 2: bilinear sampling + attention-weighted aggregation -> agg (f16).
// One wave per query; lane p owns sample point p (NHEAD*NPTS == 32 == wave32).
// ---------------------------------------------------------------------------
__global__ void __launch_bounds__(256) sample_kernel(
    const float* __restrict__ E, const float* __restrict__ locs,
    const float* __restrict__ attn, _Float16* __restrict__ agg_h,
    const int* __restrict__ Hp, const int* __restrict__ Wp_) {
  const int H = Hp[0], W = Wp_[0];
  const int lane = threadIdx.x & 31;
  const int q = blockIdx.x * 8 + (threadIdx.x >> 5);
  const int b = q >> 10;   // M = 1024
  const float* Eb = E + (size_t)b * (size_t)(H * W) * DDIM;

  // lane = point index: compute corner indices + fused weights
  float lx = locs[(size_t)q * 64 + lane * 2 + 0];
  float ly = locs[(size_t)q * 64 + lane * 2 + 1];
  float aw = attn[(size_t)q * 32 + lane] * (1.0f / NHEAD);
  float ix = ((lx + 1.f) * W - 1.f) * 0.5f;
  float iy = ((ly + 1.f) * H - 1.f) * 0.5f;
  float x0f = floorf(ix), y0f = floorf(iy);
  int   x0 = (int)x0f, y0 = (int)y0f;
  float wx1 = ix - x0f, wx0 = 1.f - wx1;
  float wy1 = iy - y0f, wy0 = 1.f - wy1;
  int   xs[2]  = {x0, x0 + 1}; int   ys[2]  = {y0, y0 + 1};
  float wxs[2] = {wx0, wx1};   float wys[2] = {wy0, wy1};
  int idx[4]; float wgt[4];
  #pragma unroll
  for (int cy = 0; cy < 2; ++cy)
    #pragma unroll
    for (int cx = 0; cx < 2; ++cx) {
      int xc = xs[cx], yc = ys[cy];
      bool valid = (xc >= 0) && (xc < W) && (yc >= 0) && (yc < H);
      int xi = min(max(xc, 0), W - 1);
      int yi = min(max(yc, 0), H - 1);
      idx[cy * 2 + cx] = yi * W + xi;
      wgt[cy * 2 + cx] = valid ? (aw * wxs[cx] * wys[cy]) : 0.f;
    }

  // lane owns channels d0..d0+7; sweep all 32 points via shuffle broadcast
  const int d0 = lane * 8;
  float4 acc0 = make_float4(0, 0, 0, 0), acc1 = make_float4(0, 0, 0, 0);
  for (int p = 0; p < 32; ++p) {
    // prefetch next point's corner rows (global_prefetch_b8)
    if (p + 1 < 32) {
      #pragma unroll
      for (int c = 0; c < 4; ++c) {
        int nidx = __shfl(idx[c], p + 1);
        __builtin_prefetch(Eb + (size_t)nidx * DDIM + d0, 0, 1);
      }
    }
    #pragma unroll
    for (int c = 0; c < 4; ++c) {
      int   ridx = __shfl(idx[c], p);
      float rw   = __shfl(wgt[c], p);
      if (rw != 0.f) {   // wave-uniform predicate
        const float4* s = (const float4*)(Eb + (size_t)ridx * DDIM + d0);
        float4 v0 = s[0], v1 = s[1];
        acc0.x += v0.x * rw; acc0.y += v0.y * rw; acc0.z += v0.z * rw; acc0.w += v0.w * rw;
        acc1.x += v1.x * rw; acc1.y += v1.y * rw; acc1.z += v1.z * rw; acc1.w += v1.w * rw;
      }
    }
  }
  v8h h;
  h[0] = (_Float16)acc0.x; h[1] = (_Float16)acc0.y;
  h[2] = (_Float16)acc0.z; h[3] = (_Float16)acc0.w;
  h[4] = (_Float16)acc1.x; h[5] = (_Float16)acc1.y;
  h[6] = (_Float16)acc1.z; h[7] = (_Float16)acc1.w;
  *(v8h*)(agg_h + (size_t)q * DDIM + d0) = h;
}

// ---------------------------------------------------------------------------
// Kernel 3: out = agg @ Wp + bp   (8192 x 256 x 256)
// 8 waves/block; block does a 16x256 output tile; A-tile staged via TDM.
// ---------------------------------------------------------------------------
__global__ void __launch_bounds__(256) out_gemm_kernel(
    const _Float16* __restrict__ agg_h, int totalQ,
    const _Float16* __restrict__ wpP,
    const float* __restrict__ bp, float* __restrict__ out) {
  __shared__ _Float16 ldsA[16 * DDIM];
  const int tid  = threadIdx.x;
  const int lane = tid & 31;
  const int wave = tid >> 5;
  const int q0   = blockIdx.x * 16;

  if (wave == 0) {
    tdm_copy_f16_1d(agg_h + (size_t)q0 * DDIM, ldsA, 16 * DDIM, (unsigned)totalQ * DDIM);
    __builtin_amdgcn_s_wait_tensorcnt(0);
  }
  __syncthreads();

  for (int tt = 0; tt < 2; ++tt) {
    const int nt = wave * 2 + tt;
    v8f acc = {};
    for (int ks = 0; ks < KSTEPS; ++ks) {
      v16h a = load_a_frag_lds(ldsA, lane, ks);
      v16h b = load_b_frag(wpP, 16, ks, nt, lane);
      acc = wmma_f16(a, b, acc);
    }
    const int col  = nt * 16 + (lane & 15);
    const float bs = bp[col];
    #pragma unroll
    for (int v = 0; v < 8; ++v) {
      int row = v + 8 * (lane >> 4);
      out[(size_t)(q0 + row) * DDIM + col] = acc[v] + bs;
    }
  }
}

// ---------------------------------------------------------------------------
// Launch
// ---------------------------------------------------------------------------
extern "C" void kernel_launch(void* const* d_in, const int* in_sizes, int n_in,
                              void* d_out, int out_size, void* d_ws, size_t ws_size,
                              hipStream_t stream) {
  const float* SADQ = (const float*)d_in[0];
  const float* E    = (const float*)d_in[1];
  const float* Wo   = (const float*)d_in[2];
  const float* bo   = (const float*)d_in[3];
  const float* Wa   = (const float*)d_in[4];
  const float* ba   = (const float*)d_in[5];
  const float* Wr   = (const float*)d_in[6];
  const float* br   = (const float*)d_in[7];
  const float* Wp   = (const float*)d_in[8];
  const float* bp   = (const float*)d_in[9];
  const int*   Hp   = (const int*)d_in[10];
  const int*   Wdp  = (const int*)d_in[11];
  float* out = (float*)d_out;

  const int totalQ = in_sizes[0] / DDIM;  // B*M = 8192

  // workspace layout (bytes)
  char* ws = (char*)d_ws;
  _Float16* woP    = (_Float16*)(ws + 0);        //  32 KB
  _Float16* waP    = (_Float16*)(ws + 32768);    //  16 KB
  _Float16* wpP    = (_Float16*)(ws + 49152);    // 128 KB -> 180224
  _Float16* sadqH  = (_Float16*)(ws + 180224);   //   4 MB -> 4374528
  float*    attnBuf= (float*)   (ws + 4374528);  //   1 MB -> 5423104
  float*    locsBuf= (float*)   (ws + 5423104);  //   2 MB -> 7520256
  _Float16* aggH   = (_Float16*)(ws + 7520256);  //   4 MB -> 11714560

  pack_weights_kernel<<<(DDIM * 64  + 255) / 256, 256, 0, stream>>>(Wo, woP, 64);
  pack_weights_kernel<<<(DDIM * 32  + 255) / 256, 256, 0, stream>>>(Wa, waP, 32);
  pack_weights_kernel<<<(DDIM * 256 + 255) / 256, 256, 0, stream>>>(Wp, wpP, 256);

  const int n4 = totalQ * DDIM / 4;
  f32_to_f16_kernel<<<(n4 + 255) / 256, 256, 0, stream>>>(SADQ, sadqH, n4);

  proj_kernel<<<totalQ / 16, 32, 0, stream>>>(sadqH, totalQ, woP, waP, bo, ba, Wr, br,
                                              attnBuf, locsBuf);
  sample_kernel<<<totalQ / 8, 256, 0, stream>>>(E, locsBuf, attnBuf, aggH, Hp, Wdp);
  out_gemm_kernel<<<totalQ / 16, 256, 0, stream>>>(aggH, totalQ, wpP, bp, out);
}
